// AE_GCN_5488968205035
// MI455X (gfx1250) — compile-verified, matrix operand order checked by reading, not run on hardware
//
#include <hip/hip_runtime.h>

typedef __attribute__((ext_vector_type(16))) _Float16 v16h;
typedef __attribute__((ext_vector_type(8)))  _Float16 v8h;
typedef __attribute__((ext_vector_type(4)))  _Float16 v4h;
typedef __attribute__((ext_vector_type(8)))  float    v8f;

namespace {
constexpr int NN  = 100000;   // nodes
constexpr int EE  = 1600000;  // edges
constexpr int NIN = 500;
constexpr int KCL = 10;
constexpr int NZ  = 10;
inline int kpad(int k) { return (k + 31) & ~31; }
}

// ---------------------------------------------------------------------------
// WMMA GEMM, all-f16 data path, double-buffered LDS pipeline:
//   C = act(A[M x K] @ W[K x Nc] + bias)
// A:  f16 row-major, lda = kpad(K), K-pad columns zero.
// Bt: f16 TRANSPOSED weights [Nc][kpad(K)], K-pad zero (prepped once).
// C16: f16 out, ldc = kpad(Nc), pad columns zeroed. C32: optional f32 out.
// Block tile 128x128, K-step 32, 8 wave32 waves, 2x4 accumulators per wave.
// Per K-step: prefetch t+1 (4x global b128 -> regs) overlaps 8 WMMAs on t,
// then 4x ds_store_b128 to the alternate buffer; one barrier per step.
// ---------------------------------------------------------------------------
__global__ __launch_bounds__(256) void gemm_f16_wmma(
    const _Float16* __restrict__ A, int lda,
    const _Float16* __restrict__ Bt, int ldb,
    const float* __restrict__ bias,
    _Float16* __restrict__ C16, int ldc,
    float* __restrict__ C32,
    int M, int K, int Nc, int do_relu)
{
    __shared__ __align__(16) _Float16 As[2][128][40];  // pitch 80B: 16B-aligned rows
    __shared__ __align__(16) _Float16 Bs[2][128][40];

    const int tid  = threadIdx.x;
    const int lane = tid & 31;
    const int wave = tid >> 5;
    const int lrow = lane & 15;
    const int hi   = lane >> 4;

    const int m0 = blockIdx.y * 128;
    const int n0 = blockIdx.x * 128;

    const int rt0 = (wave & 3) * 2;     // row tiles rt0, rt0+1
    const int ct0 = (wave >> 2) * 4;    // col tiles ct0 .. ct0+3

    // staging assignment: thread -> (row 0..127, 16-half chunk 0/1)
    const int srow = tid >> 1;
    const int skc  = (tid & 1) * 16;
    const int ga   = m0 + srow < M  ? m0 + srow : M - 1;   // clamped rows: never stored
    const int gb   = n0 + srow < Nc ? n0 + srow : Nc - 1;
    const _Float16* aBase = A  + (size_t)ga * lda + skc;
    const _Float16* bBase = Bt + (size_t)gb * ldb + skc;

    v8f acc[2][4] = {};

    const int kt = (K + 31) >> 5;

    // prologue: stage tile 0 into buffer 0
    {
        const v8h* ap = (const v8h*)aBase;
        const v8h* bp = (const v8h*)bBase;
        *(v8h*)&As[0][srow][skc]     = ap[0];
        *(v8h*)&As[0][srow][skc + 8] = ap[1];
        *(v8h*)&Bs[0][srow][skc]     = bp[0];
        *(v8h*)&Bs[0][srow][skc + 8] = bp[1];
    }
    __syncthreads();

    for (int t = 0; t < kt; ++t) {
        const int cur = t & 1;
        const int nxt = cur ^ 1;
        // prefetch next K-tile into registers (clamped on last iter; redundant store)
        const int nk = ((t + 1 < kt) ? (t + 1) : t) << 5;
        const v8h* ap = (const v8h*)(aBase + nk);
        const v8h* bp = (const v8h*)(bBase + nk);
        const v8h na0 = ap[0];
        const v8h na1 = ap[1];
        const v8h nb0 = bp[0];
        const v8h nb1 = bp[1];

        // fragments from current buffer
        v16h af[2], bf[4];
        #pragma unroll
        for (int i = 0; i < 2; ++i) {
            const int rw = (rt0 + i) * 16 + lrow;
            v8h lo = *(const v8h*)&As[cur][rw][hi * 8];
            v8h hh = *(const v8h*)&As[cur][rw][16 + hi * 8];
            af[i] = __builtin_shufflevector(lo, hh, 0,1,2,3,4,5,6,7,8,9,10,11,12,13,14,15);
        }
        #pragma unroll
        for (int j = 0; j < 4; ++j) {
            const int cw = (ct0 + j) * 16 + lrow;
            v8h lo = *(const v8h*)&Bs[cur][cw][hi * 8];
            v8h hh = *(const v8h*)&Bs[cur][cw][16 + hi * 8];
            bf[j] = __builtin_shufflevector(lo, hh, 0,1,2,3,4,5,6,7,8,9,10,11,12,13,14,15);
        }
        #pragma unroll
        for (int i = 0; i < 2; ++i)
            #pragma unroll
            for (int j = 0; j < 4; ++j)
                acc[i][j] = __builtin_amdgcn_wmma_f32_16x16x32_f16(
                    false, af[i], false, bf[j], (short)0, acc[i][j], false, false);

        // stage next tile into alternate buffer (reads of it finished last iter)
        *(v8h*)&As[nxt][srow][skc]     = na0;
        *(v8h*)&As[nxt][srow][skc + 8] = na1;
        *(v8h*)&Bs[nxt][srow][skc]     = nb0;
        *(v8h*)&Bs[nxt][srow][skc + 8] = nb1;
        __syncthreads();
    }

    // epilogue: C layout VGPR r -> M = hi*8 + r, N = lane&15
    #pragma unroll
    for (int j = 0; j < 4; ++j) {
        const int col = n0 + (ct0 + j) * 16 + lrow;
        const float bv = (bias != nullptr && col < Nc) ? bias[col] : 0.f;
        #pragma unroll
        for (int i = 0; i < 2; ++i) {
            #pragma unroll
            for (int r = 0; r < 8; ++r) {
                const int row = m0 + (rt0 + i) * 16 + hi * 8 + r;
                if (row >= M) continue;
                if (col < Nc) {
                    float v = acc[i][j][r] + bv;
                    if (do_relu) v = fmaxf(v, 0.f);
                    if (C16) C16[(size_t)row * ldc + col] = (_Float16)v;
                    if (C32) C32[(size_t)row * Nc + col] = v;
                } else if (col < ldc) {
                    if (C16) C16[(size_t)row * ldc + col] = (_Float16)0.f;  // K-pad zeros
                }
            }
        }
    }
}

// ---------------------------------------------------------------------------
// SpMM scatter: out[dst[e], :] += w[e] * H16[src[e], :]  (out pre-zeroed f32)
// Block = 64 edges x 4 threads; edge indices read once; columns looped.
// ---------------------------------------------------------------------------
__global__ __launch_bounds__(256) void spmm_scatter(
    const _Float16* __restrict__ H, int ldh,
    const int* __restrict__ src, const int* __restrict__ dst,
    const float* __restrict__ ew, float* __restrict__ out, int E, int F)
{
    const int e = blockIdx.x * 64 + (threadIdx.x >> 2);
    if (e >= E) return;
    const int q = threadIdx.x & 3;
    const int s = src[e];
    const int d = dst[e];
    const float w = ew[e];
    const _Float16* hrow = H + (size_t)s * ldh;
    float* orow = out + (size_t)d * F;
    for (int c0 = q * 4; c0 < F; c0 += 16) {
        if (c0 + 4 <= F) {
            v4h h4 = *(const v4h*)(hrow + c0);   // 8B aligned: ldh%32==0, c0%4==0
            #pragma unroll
            for (int i = 0; i < 4; ++i)
                atomicAdd(&orow[c0 + i], w * (float)h4[i]);
        } else {
            for (int c = c0; c < F; ++c)
                atomicAdd(&orow[c], w * (float)hrow[c]);
        }
    }
}

// out16 = 0.5*relu(agg) + 0.5*tra16, zero in K-pad columns (stride KP)
__global__ __launch_bounds__(256) void relu_blend_f16(
    const float* __restrict__ agg, const _Float16* __restrict__ tra,
    _Float16* __restrict__ out, int rows, int F, int KP)
{
    long long i = (long long)blockIdx.x * 256 + threadIdx.x;
    if (i >= (long long)rows * KP) return;
    const int r = (int)(i / KP);
    const int c = (int)(i % KP);
    _Float16 v = (_Float16)0.f;
    if (c < F)
        v = (_Float16)(0.5f * fmaxf(agg[(size_t)r * F + c], 0.f) + 0.5f * (float)tra[i]);
    out[i] = v;
}

// f32 [rows x F] -> f16 [rows x KP] with zero pad (used once for x)
__global__ __launch_bounds__(256) void cvt_pad_f16(
    const float* __restrict__ in, _Float16* __restrict__ out, int rows, int F, int KP)
{
    long long i = (long long)blockIdx.x * 256 + threadIdx.x;
    if (i >= (long long)rows * KP) return;
    const int r = (int)(i / KP);
    const int c = (int)(i % KP);
    out[i] = (c < F) ? (_Float16)in[(size_t)r * F + c] : (_Float16)0.f;
}

// W[K x Nc] f32 -> Wt[Nc x KP] f16 transposed, zero pad (once per weight)
__global__ __launch_bounds__(256) void prep_wt(
    const float* __restrict__ W, _Float16* __restrict__ Wt, int K, int Nc, int KP)
{
    long long i = (long long)blockIdx.x * 256 + threadIdx.x;
    if (i >= (long long)Nc * KP) return;
    const int n = (int)(i / KP);
    const int k = (int)(i % KP);
    Wt[i] = (k < K) ? (_Float16)W[(size_t)k * Nc + n] : (_Float16)0.f;
}

// row softmax over K=10 (f32 in, stride 10)
__global__ __launch_bounds__(256) void softmax10(
    const float* __restrict__ in, float* __restrict__ out, int n)
{
    int i = blockIdx.x * 256 + threadIdx.x;
    if (i >= n) return;
    float v[KCL];
    float m = -1e30f;
    #pragma unroll
    for (int k = 0; k < KCL; ++k) { v[k] = in[(size_t)i * KCL + k]; m = fmaxf(m, v[k]); }
    float ssum = 0.f;
    #pragma unroll
    for (int k = 0; k < KCL; ++k) { v[k] = __expf(v[k] - m); ssum += v[k]; }
    float inv = 1.f / ssum;
    #pragma unroll
    for (int k = 0; k < KCL; ++k) out[(size_t)i * KCL + k] = v[k] * inv;
}

// Student-t, V=1: q_k = (1+dist_k)^-1 normalized (z f32 from d_out)
__global__ __launch_bounds__(256) void student_q(
    const float* __restrict__ z, const float* __restrict__ cluster,
    float* __restrict__ q, int n)
{
    int i = blockIdx.x * 256 + threadIdx.x;
    if (i >= n) return;
    float zz[NZ];
    #pragma unroll
    for (int j = 0; j < NZ; ++j) zz[j] = z[(size_t)i * NZ + j];
    float qv[KCL];
    float ssum = 0.f;
    #pragma unroll
    for (int k = 0; k < KCL; ++k) {
        float dist = 0.f;
        #pragma unroll
        for (int j = 0; j < NZ; ++j) {
            float d = zz[j] - cluster[k * NZ + j];
            dist += d * d;
        }
        float t = 1.0f / (1.0f + dist);
        qv[k] = t;
        ssum += t;
    }
    float inv = 1.f / ssum;
    #pragma unroll
    for (int k = 0; k < KCL; ++k) q[(size_t)i * KCL + k] = qv[k] * inv;
}

// ---------------------------------------------------------------------------
// Host orchestration
// ---------------------------------------------------------------------------
static inline void launch_gemm(const _Float16* A, int lda, const _Float16* Bt, int ldb,
                               const float* bias, _Float16* C16, int ldc, float* C32,
                               int M, int K, int Nc, int relu, hipStream_t s)
{
    dim3 grid((Nc + 127) / 128, (M + 127) / 128);
    gemm_f16_wmma<<<grid, dim3(256), 0, s>>>(A, lda, Bt, ldb, bias, C16, ldc, C32,
                                             M, K, Nc, relu);
}

static inline void launch_spmm(const _Float16* H, int ldh, const int* src, const int* dst,
                               const float* ew, float* out, int F, hipStream_t s)
{
    hipMemsetAsync((void*)out, 0, (size_t)NN * F * sizeof(float), s);
    dim3 grid((EE + 63) / 64);
    spmm_scatter<<<grid, dim3(256), 0, s>>>(H, ldh, src, dst, ew, out, EE, F);
}

static inline void launch_blend(const float* agg, const _Float16* tra, _Float16* out,
                                int F, int KP, hipStream_t s)
{
    long long n = (long long)NN * KP;
    relu_blend_f16<<<dim3((unsigned)((n + 255) / 256)), dim3(256), 0, s>>>(
        agg, tra, out, NN, F, KP);
}

static inline void launch_prep_wt(const float* W, _Float16* Wt, int K, int Nc,
                                  hipStream_t s)
{
    const int KP = kpad(K);
    long long n = (long long)Nc * KP;
    prep_wt<<<dim3((unsigned)((n + 255) / 256)), dim3(256), 0, s>>>(W, Wt, K, Nc, KP);
}

extern "C" void kernel_launch(void* const* d_in, const int* in_sizes, int n_in,
                              void* d_out, int out_size, void* d_ws, size_t ws_size,
                              hipStream_t stream)
{
    (void)in_sizes; (void)n_in; (void)out_size; (void)ws_size;
    const float* x       = (const float*)d_in[0];
    const float* ew      = (const float*)d_in[1];
    const float* w_enc1  = (const float*)d_in[2];  const float* b_enc1 = (const float*)d_in[3];
    const float* w_enc2  = (const float*)d_in[4];  const float* b_enc2 = (const float*)d_in[5];
    const float* w_enc3  = (const float*)d_in[6];  const float* b_enc3 = (const float*)d_in[7];
    const float* w_z     = (const float*)d_in[8];  const float* b_z    = (const float*)d_in[9];
    const float* w_dec1  = (const float*)d_in[10]; const float* b_dec1 = (const float*)d_in[11];
    const float* w_dec2  = (const float*)d_in[12]; const float* b_dec2 = (const float*)d_in[13];
    const float* w_dec3  = (const float*)d_in[14]; const float* b_dec3 = (const float*)d_in[15];
    const float* w_xbar  = (const float*)d_in[16]; const float* b_xbar = (const float*)d_in[17];
    const float* gw1     = (const float*)d_in[18];
    const float* gw2     = (const float*)d_in[19];
    const float* gw3     = (const float*)d_in[20];
    const float* gw4     = (const float*)d_in[21];
    const float* gw5     = (const float*)d_in[22];
    const float* cluster = (const float*)d_in[23];
    const int*   esrc    = (const int*)d_in[24];
    const int*   edst    = (const int*)d_in[25];

    float* out = (float*)d_out;
    const size_t N = NN;
    float* xbar_o = out;                                  // [N*500]
    float* q_o    = out + N * NIN;                        // [N*10]
    float* pred_o = out + N * NIN + N * KCL;              // [N*10]
    float* z_o    = out + N * NIN + N * KCL + N * KCL;    // [N*10]

    // --- workspace carving (64B aligned chunks) ---
    char* wsb = (char*)d_ws;
    size_t off = 0;
    auto carve = [&](size_t bytes) -> char* {
        char* p = wsb + off;
        off = (off + bytes + 63) & ~(size_t)63;
        return p;
    };
    const int KP500 = 512, KP2000 = 2016, KP10 = 32;
    // f32 aggregation buffers
    float* AGG_B = (float*)carve(N * 2000 * 4);
    float* AGG_M = (float*)carve(N * 500 * 4);
    float* AGG_S = (float*)carve(N * 10 * 4);
    // f16 activations (K-padded strides)
    _Float16* X16 = (_Float16*)carve(N * KP500 * 2);
    _Float16* T1  = (_Float16*)carve(N * KP500 * 2);
    _Float16* T2  = (_Float16*)carve(N * KP500 * 2);
    _Float16* T3  = (_Float16*)carve(N * KP2000 * 2);
    _Float16* Z16 = (_Float16*)carve(N * KP10 * 2);
    _Float16* D1  = (_Float16*)carve(N * KP2000 * 2);   // d1; reused for g3 out
    _Float16* M0b = (_Float16*)carve(N * KP500 * 2);    // d2; g1/g2 out
    _Float16* M1b = (_Float16*)carve(N * KP500 * 2);    // d3
    _Float16* S0  = (_Float16*)carve(N * KP10 * 2);     // g4/g5 out
    _Float16* S2  = (_Float16*)carve(N * KP10 * 2);     // blend4 out
    // f16 transposed weights [Nc][kpad(K)]
    _Float16* wt_enc1 = (_Float16*)carve((size_t)500 * KP500 * 2);
    _Float16* wt_enc2 = (_Float16*)carve((size_t)500 * KP500 * 2);
    _Float16* wt_enc3 = (_Float16*)carve((size_t)2000 * KP500 * 2);
    _Float16* wt_z    = (_Float16*)carve((size_t)10 * KP2000 * 2);
    _Float16* wt_dec1 = (_Float16*)carve((size_t)2000 * KP10 * 2);
    _Float16* wt_dec2 = (_Float16*)carve((size_t)500 * KP2000 * 2);
    _Float16* wt_dec3 = (_Float16*)carve((size_t)500 * KP500 * 2);
    _Float16* wt_xbar = (_Float16*)carve((size_t)500 * KP500 * 2);
    _Float16* wt_g1   = (_Float16*)carve((size_t)500 * KP500 * 2);
    _Float16* wt_g2   = (_Float16*)carve((size_t)500 * KP500 * 2);
    _Float16* wt_g3   = (_Float16*)carve((size_t)2000 * KP500 * 2);
    _Float16* wt_g4   = (_Float16*)carve((size_t)10 * KP2000 * 2);
    _Float16* wt_g5   = (_Float16*)carve((size_t)10 * KP10 * 2);

    // --- prep: f16 conversion / transpose (cheap, once per call) ---
    {
        long long n = (long long)N * KP500;
        cvt_pad_f16<<<dim3((unsigned)((n + 255) / 256)), dim3(256), 0, stream>>>(
            x, X16, NN, NIN, KP500);
    }
    launch_prep_wt(w_enc1, wt_enc1, 500, 500,  stream);
    launch_prep_wt(w_enc2, wt_enc2, 500, 500,  stream);
    launch_prep_wt(w_enc3, wt_enc3, 500, 2000, stream);
    launch_prep_wt(w_z,    wt_z,    2000, 10,  stream);
    launch_prep_wt(w_dec1, wt_dec1, 10,  2000, stream);
    launch_prep_wt(w_dec2, wt_dec2, 2000, 500, stream);
    launch_prep_wt(w_dec3, wt_dec3, 500, 500,  stream);
    launch_prep_wt(w_xbar, wt_xbar, 500, 500,  stream);
    launch_prep_wt(gw1,    wt_g1,   500, 500,  stream);
    launch_prep_wt(gw2,    wt_g2,   500, 500,  stream);
    launch_prep_wt(gw3,    wt_g3,   500, 2000, stream);
    launch_prep_wt(gw4,    wt_g4,   2000, 10,  stream);
    launch_prep_wt(gw5,    wt_g5,   10,  10,   stream);

    // --- AE ---
    launch_gemm(X16, KP500, wt_enc1, KP500, b_enc1, T1, KP500, nullptr, NN, 500, 500, 1, stream);
    launch_gemm(T1, KP500, wt_enc2, KP500, b_enc2, T2, KP500, nullptr, NN, 500, 500, 1, stream);
    launch_gemm(T2, KP500, wt_enc3, KP500, b_enc3, T3, KP2000, nullptr, NN, 500, 2000, 1, stream);
    launch_gemm(T3, KP2000, wt_z, KP2000, b_z, Z16, KP10, z_o, NN, 2000, 10, 0, stream);
    launch_gemm(Z16, KP10, wt_dec1, KP10, b_dec1, D1, KP2000, nullptr, NN, 10, 2000, 1, stream);
    launch_gemm(D1, KP2000, wt_dec2, KP2000, b_dec2, M0b, KP500, nullptr, NN, 2000, 500, 1, stream);
    launch_gemm(M0b, KP500, wt_dec3, KP500, b_dec3, M1b, KP500, nullptr, NN, 500, 500, 1, stream);
    launch_gemm(M1b, KP500, wt_xbar, KP500, b_xbar, nullptr, KP500, xbar_o, NN, 500, 500, 0, stream);

    // --- GCN (sigma = 0.5) ---
    launch_gemm(X16, KP500, wt_g1, KP500, nullptr, M0b, KP500, nullptr, NN, 500, 500, 0, stream);
    launch_spmm(M0b, KP500, esrc, edst, ew, AGG_M, 500, stream);
    launch_blend(AGG_M, T1, T1, 500, KP500, stream);

    launch_gemm(T1, KP500, wt_g2, KP500, nullptr, M0b, KP500, nullptr, NN, 500, 500, 0, stream);
    launch_spmm(M0b, KP500, esrc, edst, ew, AGG_M, 500, stream);
    launch_blend(AGG_M, T2, T2, 500, KP500, stream);

    launch_gemm(T2, KP500, wt_g3, KP500, nullptr, D1, KP2000, nullptr, NN, 500, 2000, 0, stream);
    launch_spmm(D1, KP2000, esrc, edst, ew, AGG_B, 2000, stream);
    launch_blend(AGG_B, T3, T3, 2000, KP2000, stream);

    launch_gemm(T3, KP2000, wt_g4, KP2000, nullptr, S0, KP10, nullptr, NN, 2000, 10, 0, stream);
    launch_spmm(S0, KP10, esrc, edst, ew, AGG_S, 10, stream);
    launch_blend(AGG_S, Z16, S2, 10, KP10, stream);

    launch_gemm(S2, KP10, wt_g5, KP10, nullptr, S0, KP10, nullptr, NN, 10, 10, 0, stream);
    launch_spmm(S0, KP10, esrc, edst, ew, AGG_S, 10, stream);
    softmax10<<<dim3((NN + 255) / 256), dim3(256), 0, stream>>>(AGG_S, pred_o, NN);

    // --- Student-t q ---
    student_q<<<dim3((NN + 255) / 256), dim3(256), 0, stream>>>(z_o, cluster, q_o, NN);
}
